// PointTransformerBlock_17841294147944
// MI455X (gfx1250) — compile-verified, matrix operand order checked by reading, not run on hardware
//
#include <hip/hip_runtime.h>
#include <hip/hip_bf16.h>
#include <float.h>

#define BATCH 2
#define NPTS  8192
#define DIM   64
#define KNN   16
#define TOTAL (BATCH * NPTS)

// fragment-major f16 weight buffer: 8 frags x 32 lanes x 16 halves per matrix
#define WFRAG_HALFS (8 * 32 * 16)
// matrix slots inside the prefragmented buffer
#define WF_W1     0
#define WF_PHI    1
#define WF_PSI    2
#define WF_ALPHA  3
#define WF_GAMMA  4
#define WF_D1     5
#define WF_D2     6
#define WF_W2     7

typedef __attribute__((ext_vector_type(16))) _Float16 v16h;
typedef __attribute__((ext_vector_type(8)))  float    v8f;

// K index inside a 32-wide k-tile for half h (0..15) and lane k-group kg (0/1),
// per CDNA5 ISA 7.12.2 "16-bit A-Matrix 16x32" table.
__device__ __forceinline__ int kmap(int h, int kg) {
  return ((h >> 3) << 4) + (kg << 3) + (h & 7);
}

// A fragment (16x32 f16) from an LDS row-major [16][64] tile, k-tile kt.
// Halves 0..7 and 8..15 are K-contiguous -> compiler emits 2x ds_load_b128.
__device__ __forceinline__ v16h load_A_lds(const _Float16* tile, int lane, int kt) {
  int row = lane & 15;
  int kg  = (lane >> 4) & 1;
  v16h a;
#pragma unroll
  for (int h = 0; h < 16; ++h) a[h] = tile[row * DIM + kt * 32 + kmap(h, kg)];
  return a;
}

// B fragment from the prefragmented f16 weight buffer: one aligned 32B read.
__device__ __forceinline__ v16h load_B_pre(const _Float16* Wf, int lane, int kt, int nt) {
  return *(const v16h*)(Wf + (((kt << 2) + nt) * 32 + lane) * 16);
}

// Wave-level Y[16][64] = A[16][64] @ W^T + bias  (f16 inputs, f32 accumulate).
// Bias is seeded into the accumulator BEFORE the WMMAs (D = A*B + C), so there
// is no post-WMMA VALU (avoids WMMA->VALU hazard NOPs and runtime branches).
template <bool HAS_BIAS>
__device__ __forceinline__ void wave_gemm64(const _Float16* Atile, const _Float16* Wf,
                                            const float* bias, int lane, v8f acc[4]) {
  v16h a0 = load_A_lds(Atile, lane, 0);
  v16h a1 = load_A_lds(Atile, lane, 1);
#pragma unroll
  for (int nt = 0; nt < 4; ++nt) {
    v8f c;
    if (HAS_BIAS) {
      float bv = bias[nt * 16 + (lane & 15)];
#pragma unroll
      for (int r = 0; r < 8; ++r) c[r] = bv;
    } else {
#pragma unroll
      for (int r = 0; r < 8; ++r) c[r] = 0.0f;
    }
    v16h b0 = load_B_pre(Wf, lane, 0, nt);
    v16h b1 = load_B_pre(Wf, lane, 1, nt);
    c = __builtin_amdgcn_wmma_f32_16x16x32_f16(false, a0, false, b0, (short)0, c, false, false);
    c = __builtin_amdgcn_wmma_f32_16x16x32_f16(false, a1, false, b1, (short)0, c, false, false);
    acc[nt] = c;
  }
}

// D tile layout (ISA 7.12.2): col = lane%16, row = (lane/16)*8 + r.
__device__ __forceinline__ void store_tiles_f16(const v8f acc[4], _Float16* tile,
                                                int lane, bool relu) {
  int col = lane & 15, rb = (lane >> 4) * 8;
#pragma unroll
  for (int nt = 0; nt < 4; ++nt)
#pragma unroll
    for (int r = 0; r < 8; ++r) {
      float v = acc[nt][r];
      if (relu) v = fmaxf(v, 0.0f);
      tile[(rb + r) * DIM + nt * 16 + col] = (_Float16)v;
    }
}

__device__ __forceinline__ void store_tiles_f32g(const v8f acc[4], float* out,
                                                 long rowbase, int lane) {
  int col = lane & 15, rb = (lane >> 4) * 8;
#pragma unroll
  for (int nt = 0; nt < 4; ++nt)
#pragma unroll
    for (int r = 0; r < 8; ++r)
      out[(rowbase + rb + r) * DIM + nt * 16 + col] = acc[nt][r];
}

// ---------------------------------------------------------------------------
// Kernel 0: pre-swizzle all weight matrices into fragment-major f16 layout.
// Element (w, frag f, lane l, half h) <- W[col][k] with zero-pad past kdim.
// ---------------------------------------------------------------------------
__global__ void prep_weights_kernel(
    const float* __restrict__ W1, const float* __restrict__ Wphi,
    const float* __restrict__ Wpsi, const float* __restrict__ Walpha,
    const float* __restrict__ Wgamma, const float* __restrict__ Wd1,
    const float* __restrict__ Wd2, const float* __restrict__ W2,
    _Float16* __restrict__ out) {
  const float* Ws[8] = {W1, Wphi, Wpsi, Walpha, Wgamma, Wd1, Wd2, W2};
  const int    kd[8] = {64, 64, 64, 64, 64, 3, 64, 64};
  int i = blockIdx.x * 256 + threadIdx.x;
  if (i >= 8 * WFRAG_HALFS) return;
  int w   = i / WFRAG_HALFS;
  int rem = i % WFRAG_HALFS;
  int f = rem >> 9;          // / (32*16)
  int l = (rem >> 4) & 31;
  int h = rem & 15;
  int kt = f >> 2, nt = f & 3;
  int col = nt * 16 + (l & 15);
  int kg  = (l >> 4) & 1;
  int k   = kt * 32 + kmap(h, kg);
  float v = (k < kd[w]) ? Ws[w][col * kd[w] + k] : 0.0f;
  out[i] = (_Float16)v;
}

// ---------------------------------------------------------------------------
// Kernel 1: KNN. Block = 256 queries of one batch; LDS-tile candidates;
// per-thread register top-16 (squared distance is order-equivalent to sqrt).
// ---------------------------------------------------------------------------
__global__ void knn_kernel(const float* __restrict__ xyz, int* __restrict__ idxg) {
  __shared__ float sx[256], sy[256], sz[256];
  const int b = blockIdx.y;
  const int q = blockIdx.x * 256 + threadIdx.x;
  const long qg = (long)b * NPTS + q;
  const float qx = xyz[qg * 3 + 0], qy = xyz[qg * 3 + 1], qz = xyz[qg * 3 + 2];

  float bd[KNN];
  int   bi[KNN];
#pragma unroll
  for (int i = 0; i < KNN; ++i) { bd[i] = FLT_MAX; bi[i] = 0; }

  for (int t = 0; t < NPTS; t += 256) {
    __syncthreads();
    const long pg = (long)b * NPTS + t + threadIdx.x;
    sx[threadIdx.x] = xyz[pg * 3 + 0];
    sy[threadIdx.x] = xyz[pg * 3 + 1];
    sz[threadIdx.x] = xyz[pg * 3 + 2];
    __syncthreads();
    for (int u = 0; u < 256; ++u) {
      float dx = qx - sx[u], dy = qy - sy[u], dz = qz - sz[u];
      float d = dx * dx + dy * dy + dz * dz;
      if (d < bd[KNN - 1]) {
        bd[KNN - 1] = d; bi[KNN - 1] = t + u;
#pragma unroll
        for (int s = KNN - 1; s > 0; --s) {
          if (bd[s] < bd[s - 1]) {
            float td = bd[s]; bd[s] = bd[s - 1]; bd[s - 1] = td;
            int   ti = bi[s]; bi[s] = bi[s - 1]; bi[s - 1] = ti;
          }
        }
      }
    }
  }
#pragma unroll
  for (int i = 0; i < KNN; ++i) idxg[qg * KNN + i] = b * NPTS + bi[i];  // global row
}

// ---------------------------------------------------------------------------
// Kernel 2: per-point linears. Wave = 16 points. Chain:
//   f = X@W1^T+b1 ; phig = (f@Wphi^T+bphi)@Wg^T+bg ; psig = (f@Wpsi^T+bpsi)@Wg^T
//   alphaP = f@Walpha^T+balpha
// ---------------------------------------------------------------------------
__global__ void point_linears_kernel(
    const float* __restrict__ Xf, const _Float16* __restrict__ wf,
    const float* __restrict__ b1, const float* __restrict__ bphi,
    const float* __restrict__ bpsi, const float* __restrict__ balpha,
    const float* __restrict__ bgamma,
    float* __restrict__ phig, float* __restrict__ psig, float* __restrict__ alphaP) {
  __shared__ _Float16 lds[4][2][16 * DIM];
  const int lane = threadIdx.x & 31;
  const int wid  = threadIdx.x >> 5;
  const long base = (long)blockIdx.x * 64 + wid * 16;
  _Float16* Fb = lds[wid][0];
  _Float16* Tb = lds[wid][1];
  v8f acc[4];

  for (int i = lane; i < 16 * DIM; i += 32)
    Tb[i] = (_Float16)Xf[(base + (i >> 6)) * DIM + (i & 63)];
  __syncthreads();

  wave_gemm64<true>(Tb, wf + WF_W1 * WFRAG_HALFS, b1, lane, acc);       // f
  __syncthreads();
  store_tiles_f16(acc, Fb, lane, false);
  __syncthreads();

  wave_gemm64<true>(Fb, wf + WF_PHI * WFRAG_HALFS, bphi, lane, acc);    // phi
  __syncthreads();
  store_tiles_f16(acc, Tb, lane, false);
  __syncthreads();
  wave_gemm64<true>(Tb, wf + WF_GAMMA * WFRAG_HALFS, bgamma, lane, acc); // phig (+bgamma)
  store_tiles_f32g(acc, phig, base, lane);
  __syncthreads();

  wave_gemm64<true>(Fb, wf + WF_PSI * WFRAG_HALFS, bpsi, lane, acc);    // psi
  __syncthreads();
  store_tiles_f16(acc, Tb, lane, false);
  __syncthreads();
  wave_gemm64<false>(Tb, wf + WF_GAMMA * WFRAG_HALFS, nullptr, lane, acc); // psig
  store_tiles_f32g(acc, psig, base, lane);

  wave_gemm64<true>(Fb, wf + WF_ALPHA * WFRAG_HALFS, balpha, lane, acc); // alphaP
  store_tiles_f32g(acc, alphaP, base, lane);
}

// ---------------------------------------------------------------------------
// Kernel 3: attention core. One wave = one query point; its 16 neighbors are
// the 16 WMMA rows. diff->h (K=3 zero-padded in prefrag Wd1), h->delta (relu),
// delta->delta@Wg^T, gamma = phig_i - psig_j + deltag ; channel softmax ;
// out_pre = sum_k rho * (alphaP_j + delta).
// ---------------------------------------------------------------------------
__global__ void attention_kernel(
    const float* __restrict__ xyz, const int* __restrict__ idxg,
    const _Float16* __restrict__ wf,
    const float* __restrict__ bd1, const float* __restrict__ bd2,
    const float* __restrict__ phig, const float* __restrict__ psig,
    const float* __restrict__ alphaP, float* __restrict__ outpre) {
  __shared__ _Float16 lds_h[4][16 * DIM];
  __shared__ _Float16 lds_d[4][16 * DIM];
  const int lane = threadIdx.x & 31;
  const int wid  = threadIdx.x >> 5;
  const long n   = (long)blockIdx.x * 4 + wid;
  const int col  = lane & 15;
  const int rb   = (lane >> 4) * 8;

  const float qx = xyz[n * 3 + 0], qy = xyz[n * 3 + 1], qz = xyz[n * 3 + 2];

  // A fragment for diff rows (row = neighbor k = lane%16), K=0..2 only.
  int jrow = idxg[n * KNN + col];
  v16h a = {};
  if ((lane >> 4) == 0) {
    a[0] = (_Float16)(qx - xyz[(long)jrow * 3 + 0]);
    a[1] = (_Float16)(qy - xyz[(long)jrow * 3 + 1]);
    a[2] = (_Float16)(qz - xyz[(long)jrow * 3 + 2]);
  }

  const _Float16* wfD1 = wf + WF_D1 * WFRAG_HALFS;
  v8f acc[4];
#pragma unroll
  for (int nt = 0; nt < 4; ++nt) {               // h = diff@Wd1^T + bd1
    float bv = bd1[nt * 16 + col];
    v8f c;
#pragma unroll
    for (int r = 0; r < 8; ++r) c[r] = bv;       // bias seeded as C
    v16h b0 = load_B_pre(wfD1, lane, 0, nt);
    c = __builtin_amdgcn_wmma_f32_16x16x32_f16(false, a, false, b0, (short)0, c, false, false);
    acc[nt] = c;
  }
  store_tiles_f16(acc, lds_h[wid], lane, false);
  __syncthreads();

  wave_gemm64<true>(lds_h[wid], wf + WF_D2 * WFRAG_HALFS, bd2, lane, acc); // pre-relu delta
  __syncthreads();
  store_tiles_f16(acc, lds_d[wid], lane, true);   // relu applied here
  __syncthreads();

  wave_gemm64<false>(lds_d[wid], wf + WF_GAMMA * WFRAG_HALFS, nullptr, lane, acc); // deltag

  // gamma = deltag + phig_i - psig_j   (bgamma already folded into phig)
  int jr[8];
#pragma unroll
  for (int r = 0; r < 8; ++r) jr[r] = idxg[n * KNN + rb + r];
#pragma unroll
  for (int nt = 0; nt < 4; ++nt) {
    int ch = nt * 16 + col;
    float pv = phig[n * DIM + ch];
#pragma unroll
    for (int r = 0; r < 8; ++r)
      acc[nt][r] = acc[nt][r] + pv - psig[(long)jr[r] * DIM + ch];
  }

  // softmax over the 64 channels of each row (row = neighbor k).
#pragma unroll
  for (int r = 0; r < 8; ++r) {
    float mx = fmaxf(fmaxf(acc[0][r], acc[1][r]), fmaxf(acc[2][r], acc[3][r]));
#pragma unroll
    for (int s = 1; s < 16; s <<= 1) mx = fmaxf(mx, __shfl_xor(mx, s, 32));
    float e0 = __expf(acc[0][r] - mx), e1 = __expf(acc[1][r] - mx);
    float e2 = __expf(acc[2][r] - mx), e3 = __expf(acc[3][r] - mx);
    float sum = e0 + e1 + e2 + e3;
#pragma unroll
    for (int s = 1; s < 16; s <<= 1) sum += __shfl_xor(sum, s, 32);
    float inv = 1.0f / sum;
    acc[0][r] = e0 * inv; acc[1][r] = e1 * inv;
    acc[2][r] = e2 * inv; acc[3][r] = e3 * inv;
  }

  // out_pre[ch] = sum_k rho[k][ch] * (alphaP_j[ch] + delta[k][ch])
  float accum[4] = {0.f, 0.f, 0.f, 0.f};
#pragma unroll
  for (int nt = 0; nt < 4; ++nt) {
    int ch = nt * 16 + col;
#pragma unroll
    for (int r = 0; r < 8; ++r) {
      float dv = (float)lds_d[wid][(rb + r) * DIM + ch];
      float av = alphaP[(long)jr[r] * DIM + ch] + dv;
      accum[nt] += acc[nt][r] * av;
    }
  }
#pragma unroll
  for (int nt = 0; nt < 4; ++nt) accum[nt] += __shfl_xor(accum[nt], 16, 32);
  if (lane < 16) {
#pragma unroll
    for (int nt = 0; nt < 4; ++nt) outpre[n * DIM + nt * 16 + lane] = accum[nt];
  }
}

// ---------------------------------------------------------------------------
// Kernel 4: out = outpre@W2^T + b2 + X   (residual)
// ---------------------------------------------------------------------------
__global__ void final_kernel(const float* __restrict__ outpre,
                             const _Float16* __restrict__ wf,
                             const float* __restrict__ b2,
                             const float* __restrict__ Xf, float* __restrict__ out) {
  __shared__ _Float16 lds[4][16 * DIM];
  const int lane = threadIdx.x & 31;
  const int wid  = threadIdx.x >> 5;
  const long base = (long)blockIdx.x * 64 + wid * 16;
  for (int i = lane; i < 16 * DIM; i += 32)
    lds[wid][i] = (_Float16)outpre[(base + (i >> 6)) * DIM + (i & 63)];
  __syncthreads();
  v8f acc[4];
  wave_gemm64<true>(lds[wid], wf + WF_W2 * WFRAG_HALFS, b2, lane, acc);
  const int col = lane & 15, rb = (lane >> 4) * 8;
#pragma unroll
  for (int nt = 0; nt < 4; ++nt)
#pragma unroll
    for (int r = 0; r < 8; ++r) {
      long row = base + rb + r;
      int ch = nt * 16 + col;
      out[row * DIM + ch] = acc[nt][r] + Xf[row * DIM + ch];
    }
}

__global__ void copy_xyz_kernel(const float* __restrict__ xyz, float* __restrict__ dst) {
  int i = blockIdx.x * 256 + threadIdx.x;
  if (i < TOTAL * 3) dst[i] = xyz[i];
}

// ---------------------------------------------------------------------------
extern "C" void kernel_launch(void* const* d_in, const int* in_sizes, int n_in,
                              void* d_out, int out_size, void* d_ws, size_t ws_size,
                              hipStream_t stream) {
  const float* xyz    = (const float*)d_in[0];
  const float* feats  = (const float*)d_in[1];
  // d_in[2] = k (== 16, hardcoded)
  const float* W1     = (const float*)d_in[3];
  const float* b1     = (const float*)d_in[4];
  const float* Wphi   = (const float*)d_in[5];
  const float* bphi   = (const float*)d_in[6];
  const float* Wpsi   = (const float*)d_in[7];
  const float* bpsi   = (const float*)d_in[8];
  const float* Walpha = (const float*)d_in[9];
  const float* balpha = (const float*)d_in[10];
  const float* Wgamma = (const float*)d_in[11];
  const float* bgamma = (const float*)d_in[12];
  const float* Wd1    = (const float*)d_in[13];
  const float* bd1    = (const float*)d_in[14];
  const float* Wd2    = (const float*)d_in[15];
  const float* bd2    = (const float*)d_in[16];
  const float* W2     = (const float*)d_in[17];
  const float* b2     = (const float*)d_in[18];

  char* ws = (char*)d_ws;
  int*   idxg   = (int*)ws;   ws += (size_t)TOTAL * KNN * sizeof(int);
  float* phig   = (float*)ws; ws += (size_t)TOTAL * DIM * sizeof(float);
  float* psig   = (float*)ws; ws += (size_t)TOTAL * DIM * sizeof(float);
  float* alphaP = (float*)ws; ws += (size_t)TOTAL * DIM * sizeof(float);
  float* outpre = (float*)ws; ws += (size_t)TOTAL * DIM * sizeof(float);
  _Float16* wfrag = (_Float16*)ws; ws += (size_t)8 * WFRAG_HALFS * sizeof(_Float16);

  float* out_xyz  = (float*)d_out;               // [B,N,3]
  float* out_feat = (float*)d_out + TOTAL * 3;   // [B,N,64]

  prep_weights_kernel<<<(8 * WFRAG_HALFS) / 256, 256, 0, stream>>>(
      W1, Wphi, Wpsi, Walpha, Wgamma, Wd1, Wd2, W2, wfrag);
  knn_kernel<<<dim3(NPTS / 256, BATCH), 256, 0, stream>>>(xyz, idxg);
  point_linears_kernel<<<TOTAL / 64, 128, 0, stream>>>(
      feats, wfrag, b1, bphi, bpsi, balpha, bgamma, phig, psig, alphaP);
  attention_kernel<<<TOTAL / 4, 128, 0, stream>>>(
      xyz, idxg, wfrag, bd1, bd2, phig, psig, alphaP, outpre);
  final_kernel<<<TOTAL / 64, 128, 0, stream>>>(outpre, wfrag, b2, feats, out_feat);
  copy_xyz_kernel<<<(TOTAL * 3 + 255) / 256, 256, 0, stream>>>(xyz, out_xyz);
}
